// MPNN_29411936043070
// MI455X (gfx1250) — matrix operand order repaired and checked by reading, not compiled, not run on hardware
//
#include <hip/hip_runtime.h>
#include <cstdint>

#define N_NODES  10000
#define N_EDGES  30000
#define N_GRAPHS 256
#define EDGE_C   8
#define HID      32      // edge-MLP hidden width
#define H_STRIDE 34      // 32 hidden + 1 bias-fold row (=1.0) + 1 zero pad row

typedef __attribute__((ext_vector_type(16))) __bf16       v16bf;
typedef __attribute__((ext_vector_type(2)))  __bf16       bf16x2;
typedef __attribute__((ext_vector_type(8)))  float        v8f;
typedef __attribute__((ext_vector_type(4)))  unsigned int v4u;

__device__ __forceinline__ __bf16 f2bf(float f) {
  unsigned u = __builtin_bit_cast(unsigned, f);
  u += 0x7FFFu + ((u >> 16) & 1u);                 // round-to-nearest-even
  unsigned short h = (unsigned short)(u >> 16);
  return __builtin_bit_cast(__bf16, h);
}

// ---------------- zero scratch accumulators ----------------
__global__ void k_zero(float* p, size_t n) {
  size_t i = (size_t)blockIdx.x * blockDim.x + threadIdx.x;
  size_t st = (size_t)gridDim.x * blockDim.x;
  for (; i < n; i += st) p[i] = 0.f;
}

// ---------------- per-destination edge counts ----------------
__global__ void k_edge_cnt(const int* dst, float* cnt) {
  int e = blockIdx.x * blockDim.x + threadIdx.x;
  if (e < N_EDGES) atomicAdd(&cnt[dst[e]], 1.f);
}

// ---------------- edge-MLP hidden: H[e,k]=relu(ea@W1+b1) (bf16), H[e,32]=1 ----------------
__global__ void k_hidden(const float* ea, const float* W1, const float* b1, __bf16* H) {
  int t = blockIdx.x * blockDim.x + threadIdx.x;
  if (t >= N_EDGES * H_STRIDE) return;
  int e = t / H_STRIDE, k = t % H_STRIDE;
  float v;
  if (k < HID) {
    v = b1[k];
#pragma unroll
    for (int j = 0; j < EDGE_C; ++j) v += ea[e * EDGE_C + j] * W1[j * HID + k];
    v = fmaxf(v, 0.f);
  } else v = (k == HID) ? 1.f : 0.f;
  H[(size_t)e * H_STRIDE + k] = f2bf(v);
}

// ---------------- gather source-node rows per edge (bf16) ----------------
__global__ void k_gather(const float* X, const int* src, __bf16* Xg, int IC) {
  int t = blockIdx.x * blockDim.x + threadIdx.x;
  if (t >= N_EDGES * IC) return;
  int e = t / IC, i = t % IC;
  Xg[t] = f2bf(X[(size_t)src[e] * IC + i]);
}

// ---------------- pack W2 (+b2 fold) into per-lane bf16 B-fragments ----------------
__global__ void k_packB(const float* W2, const float* b2, unsigned int* Bp,
                        int NKB, int NT, int IC, int OC, int SH) {
  int t = blockIdx.x * blockDim.x + threadIdx.x;
  int total = NKB * NT * 256;
  if (t >= total) return;
  int d = t & 7, lane = (t >> 3) & 31, f = t >> 8;
  int nt = f % NT, kb = f / NT;
  int n = nt * 16 + (lane & 15);
  int k0 = kb * 32 + ((lane & 16) ? 8 : 0) + 2 * d + ((d >= 4) ? 8 : 0);
  unsigned int pk = 0;
#pragma unroll
  for (int h = 0; h < 2; ++h) {
    int K = k0 + h;
    int khid = K >> SH;
    int i = K & (IC - 1);
    float w = 0.f;
    if (khid < HID)       w = W2[(size_t)khid * IC * OC + (size_t)i * OC + n];
    else if (khid == HID) w = b2[(size_t)i * OC + n];
    unsigned short bh = __builtin_bit_cast(unsigned short, f2bf(w));
    pk |= ((unsigned int)bh) << (16 * h);
  }
  Bp[t] = pk;
}

// ---------------- fused NNConv message GEMM ----------------
// msg[e,o] = sum_{k,i} H[e,k]*Xg[e,i]*W2'[k,i,o]. A built on the fly via v_pk_mul_bf16.
// B streamed global->LDS with double-buffered async-to-LDS copies shared by 8 waves.
template <int IC, int KHID, int OC, int CH>
__global__ void __launch_bounds__(256) k_gemm(const __bf16* __restrict__ Xgb,
                                              const __bf16* __restrict__ Hb,
                                              const unsigned int* __restrict__ Bp,
                                              const int* __restrict__ dst,
                                              float* __restrict__ agg, int ntiles) {
  constexpr int NT    = OC / 16;
  constexpr int NKB   = (IC * KHID) / 32;
  constexpr int NCH   = NKB / CH;
  constexpr int SH    = (IC == 16) ? 4 : 6;
  constexpr int STAGE = CH * NT * 1024;        // bytes per B chunk
  constexpr int G     = STAGE / (256 * 16);    // async b128 ops per thread per stage

  __shared__ __bf16 sXb[8][16 * IC];
  __shared__ __bf16 sHb[8][16 * H_STRIDE];
  __shared__ __align__(16) unsigned char sB[2 * STAGE];

  int lane = threadIdx.x & 31;
  int w    = threadIdx.x >> 5;
  int tile = blockIdx.x * 8 + w;
  bool valid = tile < ntiles;

  // per-wave A operands -> LDS (bf16, contiguous rows)
  if (valid) {
    const unsigned* gx = (const unsigned*)(Xgb + (size_t)tile * 16 * IC);
    unsigned* lx = (unsigned*)sXb[w];
    for (int i = lane; i < 16 * IC / 2; i += 32) lx[i] = gx[i];
    const unsigned* gh = (const unsigned*)(Hb + (size_t)tile * 16 * H_STRIDE);
    unsigned* lh = (unsigned*)sHb[w];
    for (int i = lane; i < 16 * H_STRIDE / 2; i += 32) lh[i] = gh[i];
  }

  unsigned sBoff = (unsigned)(uintptr_t)&sB[0];
  auto stage = [&](int ch, int buf) {
    unsigned long long gbase =
        (unsigned long long)(uintptr_t)Bp + (unsigned long long)ch * STAGE;
    unsigned lbase = sBoff + (unsigned)(buf * STAGE);
#pragma unroll
    for (int it = 0; it < G; ++it) {
      unsigned idx  = threadIdx.x + it * 256;
      unsigned loff = lbase + idx * 16;
      unsigned long long ga = gbase + (unsigned long long)idx * 16;
      asm volatile("global_load_async_to_lds_b128 %0, %1, off"
                   :: "v"(loff), "v"(ga) : "memory");
    }
  };

  stage(0, 0);

  int row = lane & 15;
  int hi  = (lane >> 4) & 1;
  v8f acc[NT] = {};

  for (int c = 0; c < NCH; ++c) {
    int nxt = (c + 1 < NCH) ? c + 1 : 0;     // dummy re-stage at the end keeps count constant
    stage(nxt, (c + 1) & 1);
    asm volatile("s_wait_asynccnt %0" :: "i"(G) : "memory");
    __syncthreads();                          // chunk c resident; X/H tiles resident

    const unsigned char* bufp = sB + (c & 1) * STAGE;
#pragma unroll
    for (int kbl = 0; kbl < CH; ++kbl) {
      int kb = c * CH + kbl;
      // A fragment: dword dd holds bf16 pair for K = kb*32 + hi*8 + 2dd (+8 if dd>=4)
      union { unsigned d[8]; v16bf v; } af;
#pragma unroll
      for (int dd = 0; dd < 8; ++dd) {
        int K = kb * 32 + hi * 8 + 2 * dd + ((dd >= 4) ? 8 : 0);
        int khid = K >> SH;
        int i = K & (IC - 1);
        __bf16 hv = sHb[w][row * H_STRIDE + khid];
        bf16x2 xv = *(const bf16x2*)&sXb[w][row * IC + i];
        bf16x2 hp = {hv, hv};
        bf16x2 pr = hp * xv;                  // v_pk_mul_bf16
        af.d[dd] = __builtin_bit_cast(unsigned, pr);
      }
#pragma unroll
      for (int nt = 0; nt < NT; ++nt) {
        const v4u* bp = (const v4u*)(bufp + (kbl * NT + nt) * 1024 + lane * 32);
        union { v4u q[2]; v16bf v; } bb;
        bb.q[0] = bp[0];
        bb.q[1] = bp[1];
        acc[nt] = __builtin_amdgcn_wmma_f32_16x16x32_bf16(
            false, af.v, false, bb.v, (short)0, acc[nt], false, false);
      }
    }
    __syncthreads();                          // before buffer (c&1) is overwritten
  }

  if (!valid) return;
  int eBase = tile * 16 + hi * 8;
  int o0 = lane & 15;
#pragma unroll
  for (int nt = 0; nt < NT; ++nt) {
#pragma unroll
    for (int r = 0; r < 8; ++r) {
      int e = eBase + r;
      int d = dst[e];
      atomicAdd(&agg[(size_t)d * OC + nt * 16 + o0], acc[nt][r]);
    }
  }
}

// ---------------- node update: agg/cnt + X@root + bias, relu ----------------
__global__ void k_update(const float* agg, const float* cnt, const float* Xin,
                         const float* root, const float* bias, float* Hout,
                         int IC, int OC) {
  int t = blockIdx.x * blockDim.x + threadIdx.x;
  if (t >= N_NODES * OC) return;
  int n = t / OC, o = t % OC;
  float c = fmaxf(cnt[n], 1.f);
  float v = agg[(size_t)n * OC + o] / c;
  for (int i = 0; i < IC; ++i) v += Xin[(size_t)n * IC + i] * root[i * OC + o];
  v += bias[o];
  Hout[t] = fmaxf(v, 0.f);
}

// ---------------- batchnorm stats (sum, sumsq per column) ----------------
__global__ void k_bn_stats(const float* h, float* sums, int OC) {
  __shared__ float ls[256], ls2[256];
  int tid = threadIdx.x;
  int o = tid % OC, sub = tid / OC, nsub = 256 / OC;
  float s = 0.f, s2 = 0.f;
  for (int n = blockIdx.x * nsub + sub; n < N_NODES; n += gridDim.x * nsub) {
    float v = h[(size_t)n * OC + o];
    s += v; s2 += v * v;
  }
  ls[tid] = s; ls2[tid] = s2;
  __syncthreads();
  if (sub == 0) {
    for (int j = 1; j < nsub; ++j) { s += ls[o + j * OC]; s2 += ls2[o + j * OC]; }
    atomicAdd(&sums[o], s);
    atomicAdd(&sums[OC + o], s2);
  }
}

__global__ void k_bn_apply(float* h, const float* sums, const float* g,
                           const float* b, int OC) {
  int t = blockIdx.x * blockDim.x + threadIdx.x;
  if (t >= N_NODES * OC) return;
  int o = t % OC;
  float mu = sums[o] / (float)N_NODES;
  float var = sums[OC + o] / (float)N_NODES - mu * mu;
  h[t] = (h[t] - mu) * rsqrtf(var + 1e-5f) * g[o] + b[o];
}

// ---------------- global mean pool ----------------
__global__ void k_pool(const float* h2, const int* batch, float* pool, float* gcnt) {
  int t = blockIdx.x * blockDim.x + threadIdx.x;
  if (t < N_NODES) atomicAdd(&gcnt[batch[t]], 1.f);
  if (t >= N_NODES * 128) return;
  int n = t / 128, o = t % 128;
  atomicAdd(&pool[(size_t)batch[n] * 128 + o], h2[t]);
}

// ---------------- MLP head: relu(g@lin1+b1)@lin2+b2 ----------------
__global__ void k_head(const float* pool, const float* gcnt,
                       const float* W1, const float* b1,
                       const float* W2, const float* b2, float* out) {
  int g = threadIdx.x;
  if (g >= N_GRAPHS) return;
  float c = fmaxf(gcnt[g], 1.f);
  float gv[128];
  for (int o = 0; o < 128; ++o) gv[o] = pool[g * 128 + o] / c;
  float acc = b2[0];
  for (int j = 0; j < 64; ++j) {
    float s = b1[j];
    for (int o = 0; o < 128; ++o) s += gv[o] * W1[o * 64 + j];
    acc += fmaxf(s, 0.f) * W2[j];
  }
  out[g] = acc;
}

extern "C" void kernel_launch(void* const* d_in, const int* in_sizes, int n_in,
                              void* d_out, int out_size, void* d_ws, size_t ws_size,
                              hipStream_t stream) {
  const float* x      = (const float*)d_in[0];
  const int*   ei     = (const int*)d_in[1];
  const float* ea     = (const float*)d_in[2];
  const int*   batch  = (const int*)d_in[3];
  const float* nn1_W1 = (const float*)d_in[4],  *nn1_b1 = (const float*)d_in[5];
  const float* nn1_W2 = (const float*)d_in[6],  *nn1_b2 = (const float*)d_in[7];
  const float* root1  = (const float*)d_in[8],  *bias1  = (const float*)d_in[9];
  const float* bn1_g  = (const float*)d_in[10], *bn1_b  = (const float*)d_in[11];
  const float* nn2_W1 = (const float*)d_in[12], *nn2_b1 = (const float*)d_in[13];
  const float* nn2_W2 = (const float*)d_in[14], *nn2_b2 = (const float*)d_in[15];
  const float* root2  = (const float*)d_in[16], *bias2  = (const float*)d_in[17];
  const float* bn2_g  = (const float*)d_in[18], *bn2_b  = (const float*)d_in[19];
  const float* lin1_W = (const float*)d_in[20], *lin1_b = (const float*)d_in[21];
  const float* lin2_W = (const float*)d_in[22], *lin2_b = (const float*)d_in[23];
  const int* src = ei;
  const int* dst = ei + N_EDGES;

  float* ws = (float*)d_ws;
  // workspace layout (float units); all offsets multiples of 8 -> 32B aligned
  size_t o_cnt  = 0;
  size_t o_gcnt = o_cnt  + N_NODES;
  size_t o_agg1 = o_gcnt + N_GRAPHS;
  size_t o_agg2 = o_agg1 + (size_t)N_NODES * 64;
  size_t o_bns1 = o_agg2 + (size_t)N_NODES * 128;
  size_t o_bns2 = o_bns1 + 128;
  size_t o_pool = o_bns2 + 256;
  size_t zlen   = o_pool + (size_t)N_GRAPHS * 128;
  size_t o_H    = zlen;                                     // bf16: E*34 -> E*17 floats
  size_t o_Xg1  = o_H   + (size_t)N_EDGES * H_STRIDE / 2;
  size_t o_Xg2  = o_Xg1 + (size_t)N_EDGES * 16 / 2;
  size_t o_h1   = o_Xg2 + (size_t)N_EDGES * 64 / 2;
  size_t o_h2   = o_h1  + (size_t)N_NODES * 64;
  size_t o_Bp1  = o_h2  + (size_t)N_NODES * 128;
  size_t o_Bp2  = o_Bp1 + (size_t)17 * 4 * 256;

  dim3 B(256);
  auto gridFor = [](size_t n) { return dim3((unsigned)((n + 255) / 256)); };
  const int ntiles = N_EDGES / 16;  // 1875

  k_zero<<<dim3(2048), B, 0, stream>>>(ws, zlen);
  k_edge_cnt<<<gridFor(N_EDGES), B, 0, stream>>>(dst, ws + o_cnt);
  k_packB<<<gridFor(17 * 4 * 256), B, 0, stream>>>(
      nn1_W2, nn1_b2, (unsigned int*)(ws + o_Bp1), 17, 4, 16, 64, 4);
  k_packB<<<gridFor(66 * 8 * 256), B, 0, stream>>>(
      nn2_W2, nn2_b2, (unsigned int*)(ws + o_Bp2), 66, 8, 64, 128, 6);

  // ---- conv1: 16 -> 64 ----
  k_hidden<<<gridFor((size_t)N_EDGES * H_STRIDE), B, 0, stream>>>(ea, nn1_W1, nn1_b1, (__bf16*)(ws + o_H));
  k_gather<<<gridFor((size_t)N_EDGES * 16), B, 0, stream>>>(x, src, (__bf16*)(ws + o_Xg1), 16);
  k_gemm<16, 34, 64, 1><<<dim3((ntiles + 7) / 8), B, 0, stream>>>(
      (const __bf16*)(ws + o_Xg1), (const __bf16*)(ws + o_H),
      (unsigned int*)(ws + o_Bp1), dst, ws + o_agg1, ntiles);
  k_update<<<gridFor((size_t)N_NODES * 64), B, 0, stream>>>(
      ws + o_agg1, ws + o_cnt, x, root1, bias1, ws + o_h1, 16, 64);
  k_bn_stats<<<dim3(40), B, 0, stream>>>(ws + o_h1, ws + o_bns1, 64);
  k_bn_apply<<<gridFor((size_t)N_NODES * 64), B, 0, stream>>>(ws + o_h1, ws + o_bns1, bn1_g, bn1_b, 64);

  // ---- conv2: 64 -> 128 ----
  k_hidden<<<gridFor((size_t)N_EDGES * H_STRIDE), B, 0, stream>>>(ea, nn2_W1, nn2_b1, (__bf16*)(ws + o_H));
  k_gather<<<gridFor((size_t)N_EDGES * 64), B, 0, stream>>>(ws + o_h1, src, (__bf16*)(ws + o_Xg2), 64);
  k_gemm<64, 33, 128, 2><<<dim3((ntiles + 7) / 8), B, 0, stream>>>(
      (const __bf16*)(ws + o_Xg2), (const __bf16*)(ws + o_H),
      (unsigned int*)(ws + o_Bp2), dst, ws + o_agg2, ntiles);
  k_update<<<gridFor((size_t)N_NODES * 128), B, 0, stream>>>(
      ws + o_agg2, ws + o_cnt, ws + o_h1, root2, bias2, ws + o_h2, 64, 128);
  k_bn_stats<<<dim3(40), B, 0, stream>>>(ws + o_h2, ws + o_bns2, 128);
  k_bn_apply<<<gridFor((size_t)N_NODES * 128), B, 0, stream>>>(ws + o_h2, ws + o_bns2, bn2_g, bn2_b, 128);

  // ---- pool + head ----
  k_pool<<<gridFor((size_t)N_NODES * 128), B, 0, stream>>>(ws + o_h2, batch, ws + o_pool, ws + o_gcnt);
  k_head<<<dim3(1), B, 0, stream>>>(ws + o_pool, ws + o_gcnt, lin1_W, lin1_b, lin2_W, lin2_b, (float*)d_out);
}
